// DS4DKernel_56504589746318
// MI455X (gfx1250) — matrix operand order repaired and checked by reading, not compiled, not run on hardware
//
#include <hip/hip_runtime.h>
#include <stdint.h>

// ---------------------------------------------------------------------------
// MI455X (gfx1250) implementation of:
//   deltaA = einsum('bhl,nh->bnl', du, W); S = cumsum_l(deltaA)
//   K = (C@(A@Bvec))[h] + einsum('hn,bnl->bhl', C*Bvec, S)
// bf16 WMMA (f32 accumulate) for both contractions, f32 cumsum, TDM async
// tensor loads for bf16 operand tiles, double-buffered LDS pipeline,
// non-temporal streaming for the read-once/write-once big tensors.
// ---------------------------------------------------------------------------

#define BATCH 8
#define HDIM  1024
#define NDIM  64
#define LDIM  4096

typedef __bf16 bf16_t;
typedef bf16_t v2bf   __attribute__((ext_vector_type(2)));
typedef bf16_t v8bf   __attribute__((ext_vector_type(8)));
typedef bf16_t v16bf  __attribute__((ext_vector_type(16)));
typedef float  v2f    __attribute__((ext_vector_type(2)));
typedef float  v4f    __attribute__((ext_vector_type(4)));
typedef float  v8f    __attribute__((ext_vector_type(8)));
typedef unsigned int v4u __attribute__((ext_vector_type(4)));
typedef int    v8i    __attribute__((ext_vector_type(8)));
typedef int    v4i    __attribute__((ext_vector_type(4)));

// Build a 16-element bf16 fragment from two 16-byte LDS chunks (2x ds_load_b128).
__device__ __forceinline__ v16bf ld_frag(const bf16_t* plo, const bf16_t* phi) {
  v8bf lo = *(const v8bf*)plo;
  v8bf hi = *(const v8bf*)phi;
  return __builtin_shufflevector(lo, hi, 0,1,2,3,4,5,6,7,8,9,10,11,12,13,14,15);
}

// Tensor Data Mover: 2D tile global->LDS (D# per CDNA5 ISA).
// dims/strides in element units; data_size_code: 1 => 2-byte elements (bf16).
// pad_interval/pad_amount insert LDS row padding in hardware (bank-conflict break).
// 6-arg builtin form on this toolchain.
__device__ __forceinline__ void tdm_load_2d_to_lds(
    void* lds_dst, const void* gsrc,
    uint32_t tensor_d0, uint32_t tensor_d1, uint32_t stride0,
    uint32_t tile_d0, uint32_t tile_d1,
    uint32_t data_size_code, uint32_t pad_interval, uint32_t pad_amount,
    uint32_t pad_en)
{
  uint64_t ga  = (uint64_t)(uintptr_t)gsrc;
  uint32_t lds = (uint32_t)(uintptr_t)lds_dst;   // low 32 bits of flat = LDS byte offset
  v4u g0;
  g0[0] = 1u;                                         // count=1, user descriptor
  g0[1] = lds;                                        // lds_addr
  g0[2] = (uint32_t)ga;                               // global_addr[31:0]
  g0[3] = (uint32_t)((ga >> 32) & 0x01FFFFFFu)        // global_addr[56:32]
        | (2u << 30);                                 // type=2 ("image")
  v8i g1;
  g1[0] = (int)((data_size_code << 16) | (pad_en << 20) |
                (pad_interval << 22) | (pad_amount << 25));  // wg_mask=0
  g1[1] = (int)((tensor_d0 & 0xFFFFu) << 16);                // tensor_dim0[15:0]
  g1[2] = (int)(((tensor_d0 >> 16) & 0xFFFFu) | ((tensor_d1 & 0xFFFFu) << 16));
  g1[3] = (int)(((tensor_d1 >> 16) & 0xFFFFu) | ((tile_d0  & 0xFFFFu) << 16));
  g1[4] = (int)(tile_d1 & 0xFFFFu);                          // tile_dim2 = 0
  g1[5] = (int)stride0;                                      // tensor_dim0_stride
  g1[6] = 0;
  g1[7] = 0;
  v4i g2 = {0,0,0,0};   // 2D: dims 2/3 unused
  v4i g3 = {0,0,0,0};
  v8i g4 = {0,0,0,0,0,0,0,0};
  __builtin_amdgcn_tensor_load_to_lds(g0, g1, g2, g3, g4, 0);
}

// ---------------------------------------------------------------------------
// Kernel 0: base[h] = C[h,:] @ (A @ Bvec); CBb = bf16(C * Bvec); Wb = bf16(W)
// grid 64 x 256
// ---------------------------------------------------------------------------
__global__ __launch_bounds__(256) void k_setup(
    const float* __restrict__ C, const float* __restrict__ Bv,
    const float* __restrict__ A, const float* __restrict__ W,
    float* __restrict__ base, bf16_t* __restrict__ CBb, bf16_t* __restrict__ Wb)
{
  __shared__ float sv[NDIM];
  const int tid = threadIdx.x, blk = blockIdx.x;
  if (tid < NDIM) {
    float acc = 0.f;
    for (int j = 0; j < NDIM; ++j) acc += A[tid * NDIM + j] * Bv[j];
    sv[tid] = acc;
  }
  __syncthreads();
  if (tid < 16) {
    const int h = blk * 16 + tid;
    float acc = 0.f;
    for (int n = 0; n < NDIM; ++n) acc += C[h * NDIM + n] * sv[n];
    base[h] = acc;
  }
  #pragma unroll
  for (int j = 0; j < 4; ++j) {
    const int idx = blk * 1024 + tid * 4 + j;     // covers 64K elements
    CBb[idx] = (bf16_t)(C[idx] * Bv[idx & (NDIM - 1)]);
    Wb[idx]  = (bf16_t)W[idx];
  }
}

// ---------------------------------------------------------------------------
// Kernel 1: per (batch, 128-wide L tile): deltaA tile (64n x 128l) via
// bf16 WMMA over K=H=1024 (double-buffered pipeline: prefetch k+1 while
// computing k), then intra-tile cumsum over l; writes cumA (B,L,N) f32 and
// per-tile totals sums (B,32,N).   grid (32, 8) x 256
// ---------------------------------------------------------------------------
__global__ __launch_bounds__(256) void k_gemm1(
    const float* __restrict__ du, const bf16_t* __restrict__ Wb,
    float* __restrict__ cumA, float* __restrict__ sums)
{
  __shared__ bf16_t sA[2][64 * 40];   // W tile 64n x 32k, TDM-padded rows (80 B)
  __shared__ bf16_t sB[2][128 * 40];  // du tile transposed [l][k], padded rows
  __shared__ float  sOut[64 * 132];   // deltaA tile for cumsum epilogue
  __shared__ float  sQ[64 * 4];       // quarter sums

  const int tid  = threadIdx.x;
  const int b    = blockIdx.y;
  const int tile = blockIdx.x;
  const int l0   = tile * 128;
  const int w    = tid >> 5;          // wave id: column block
  const int lane = tid & 31;
  const int half = lane >> 4;
  const int ln   = lane & 15;

  const v8f zero = {0.f,0.f,0.f,0.f,0.f,0.f,0.f,0.f};
  v8f acc[4] = {zero, zero, zero, zero};

  const float* duT = du + (size_t)b * HDIM * LDIM + l0;   // tile column base

  // Stage du k-slab (32h x 128l) f32 -> bf16, transposed [l][k]; float4 loads,
  // non-temporal (du is streamed once; keep L2 for W/cumA).
  auto stage_du = [&](int ks, int buf) {
    const float* p = duT + (size_t)ks * 32 * LDIM;
    #pragma unroll
    for (int j = 0; j < 4; ++j) {
      const int q = tid + j * 256;    // 1024 float4 groups
      const int r = q >> 5;           // k within slab (0..31)
      const int c = (q & 31) * 4;     // l within tile
      const v4f v = __builtin_nontemporal_load((const v4f*)(p + (size_t)r * LDIM + c));
      sB[buf][(c + 0) * 40 + r] = (bf16_t)v.x;
      sB[buf][(c + 1) * 40 + r] = (bf16_t)v.y;
      sB[buf][(c + 2) * 40 + r] = (bf16_t)v.z;
      sB[buf][(c + 3) * 40 + r] = (bf16_t)v.w;
    }
  };
  // TDM: W k-slab (rows n=0..63, cols h=ks*32..+31) from (64,1024) bf16 tensor.
  // pad: 16-DW rows (code 3) + 4-DW pad (code 3) -> 80 B LDS row stride.
  auto stage_W = [&](int ks, int buf) {
    tdm_load_2d_to_lds(sA[buf], Wb + ks * 32, HDIM, NDIM, HDIM, 32, 64,
                       /*dsz=*/1, /*pad_int=*/3, /*pad_amt=*/3, /*pad_en=*/1);
  };

  // Prologue: fill buffer 0.
  stage_du(0, 0);
  if (tid < 32) {
    stage_W(0, 0);
    __builtin_amdgcn_s_wait_tensorcnt(0);
  }
  __syncthreads();

  for (int ks = 0; ks < 32; ++ks) {
    const int cur = ks & 1;
    const int nxt = cur ^ 1;
    // Prefetch next slab into the other buffer while we compute.
    if (ks + 1 < 32) {
      stage_du(ks + 1, nxt);
      if (tid < 32) stage_W(ks + 1, nxt);
    }
    // Fragments (ISA §7.12.2 layouts), then back-to-back independent WMMAs.
    const bf16_t* bp = &sB[cur][(w * 16 + ln) * 40 + half * 8];
    const v16bf bfrag = ld_frag(bp, bp + 16);
    v16bf af[4];
    #pragma unroll
    for (int rb = 0; rb < 4; ++rb) {
      const bf16_t* ap = &sA[cur][(rb * 16 + ln) * 40 + half * 8];
      af[rb] = ld_frag(ap, ap + 16);
    }
    #pragma unroll
    for (int rb = 0; rb < 4; ++rb)
      acc[rb] = __builtin_amdgcn_wmma_f32_16x16x32_bf16(
          false, af[rb], false, bfrag, (short)0, acc[rb], false, false);
    if (tid < 32) __builtin_amdgcn_s_wait_tensorcnt(0);
    __syncthreads();   // next buffer ready; current buffer reads done
  }

  // Dump D fragments: VGPR r -> row (r + 8*half), lane -> col.
  #pragma unroll
  for (int rb = 0; rb < 4; ++rb)
    #pragma unroll
    for (int r = 0; r < 8; ++r)
      sOut[(rb * 16 + r + 8 * half) * 132 + w * 16 + ln] = acc[rb][r];
  __syncthreads();

  // Intra-tile cumsum over l (4 quarters of 32 per row n), write (B,L,N).
  {
    const int g = tid >> 6;         // quarter (uniform per wave)
    const int n = tid & 63;
    float run = 0.f;
    float vals[32];
    #pragma unroll
    for (int j = 0; j < 32; ++j) { run += sOut[n * 132 + g * 32 + j]; vals[j] = run; }
    sQ[n * 4 + g] = run;
    __syncthreads();
    float p = 0.f;
    for (int gg = 0; gg < g; ++gg) p += sQ[n * 4 + gg];
    const size_t ob = ((size_t)b * LDIM + l0 + g * 32) * NDIM + n;
    #pragma unroll
    for (int j = 0; j < 32; ++j)
      cumA[ob + (size_t)j * NDIM] = vals[j] + p;   // 256B-coalesced per 64 threads
    if (g == 3) sums[((size_t)b * 32 + tile) * NDIM + n] = p + run;
  }
}

// ---------------------------------------------------------------------------
// Kernel 2: exclusive scan of 32 tile totals per (b,n) -> carry offsets.
// grid 8 x 64
// ---------------------------------------------------------------------------
__global__ void k_scan(const float* __restrict__ sums, float* __restrict__ offs)
{
  const int b = blockIdx.x, n = threadIdx.x;
  float run = 0.f;
  for (int t = 0; t < 32; ++t) {
    const size_t i = ((size_t)b * 32 + t) * NDIM + n;
    const float s = sums[i];
    offs[i] = run;
    run += s;
  }
}

// ---------------------------------------------------------------------------
// Kernel 3: K[b,h,l] = base[h] + CB(128h x 64n) @ S(64n x 128l), bf16 WMMA.
// S tile = cumA + tile carry, bf16-packed during staging (float2 loads).
// grid (32 Ltiles, 8 Htiles, 8 batch) x 256
// ---------------------------------------------------------------------------
__global__ __launch_bounds__(256) void k_gemm2(
    const float* __restrict__ cumA, const float* __restrict__ offs,
    const bf16_t* __restrict__ CBb, const float* __restrict__ base,
    float* __restrict__ Kout)
{
  __shared__ bf16_t sCB[128 * 72];  // CB tile 128h x 64n, TDM-padded (144 B rows)
  __shared__ bf16_t sS[128 * 72];   // S tile transposed [l][n], padded rows
  __shared__ float  sBase[128];

  const int tid  = threadIdx.x;
  const int b    = blockIdx.z;
  const int h0   = blockIdx.y * 128;
  const int tile = blockIdx.x;
  const int l0   = tile * 128;
  const int w    = tid >> 5;
  const int lane = tid & 31;
  const int half = lane >> 4;
  const int ln   = lane & 15;

  // TDM: CB tile is 128 contiguous 128-B rows of the (1024,64) bf16 tensor.
  // pad: 32-DW rows (code 4) + 4-DW pad (code 3) -> 144 B LDS row stride.
  if (tid < 32) {
    tdm_load_2d_to_lds(sCB, CBb + (size_t)h0 * NDIM, NDIM, HDIM, NDIM, NDIM, 128,
                       /*dsz=*/1, /*pad_int=*/4, /*pad_amt=*/3, /*pad_en=*/1);
  }
  if (tid < 128) sBase[tid] = base[h0 + tid];

  // Stage S = cumA + carry, f32 -> bf16, [l][n] layout; float2/packed-bf16.
  {
    const int n2    = (tid & 31) * 2;   // constant per thread, even
    const int lbase = tid >> 5;         // 0..7
    const v2f offv = *(const v2f*)&offs[((size_t)b * 32 + tile) * NDIM + n2];
    #pragma unroll
    for (int j = 0; j < 16; ++j) {
      const int l = lbase + 8 * j;
      const v2f v = *(const v2f*)&cumA[((size_t)b * LDIM + l0 + l) * NDIM + n2];
      v2bf pv = {(bf16_t)(v.x + offv.x), (bf16_t)(v.y + offv.y)};
      *(v2bf*)&sS[l * 72 + n2] = pv;
    }
  }
  if (tid < 32) __builtin_amdgcn_s_wait_tensorcnt(0);
  __syncthreads();

  const v8f zero = {0.f,0.f,0.f,0.f,0.f,0.f,0.f,0.f};
  v8f acc[8] = {zero, zero, zero, zero, zero, zero, zero, zero};

  #pragma unroll
  for (int kk = 0; kk < 2; ++kk) {           // K = 64 = 2 x 32
    const bf16_t* bp = &sS[(w * 16 + ln) * 72 + kk * 32 + half * 8];
    const v16bf bfrag = ld_frag(bp, bp + 16);
    v16bf af[8];
    #pragma unroll
    for (int rb = 0; rb < 8; ++rb) {
      const bf16_t* ap = &sCB[(rb * 16 + ln) * 72 + kk * 32 + half * 8];
      af[rb] = ld_frag(ap, ap + 16);
    }
    #pragma unroll
    for (int rb = 0; rb < 8; ++rb)
      acc[rb] = __builtin_amdgcn_wmma_f32_16x16x32_bf16(
          false, af[rb], false, bfrag, (short)0, acc[rb], false, false);
  }

  // Epilogue: add base[h]; non-temporal streaming stores of the 134 MB output.
  const size_t ob = ((size_t)b * HDIM + h0) * LDIM + l0 + w * 16 + ln;
  #pragma unroll
  for (int rb = 0; rb < 8; ++rb)
    #pragma unroll
    for (int r = 0; r < 8; ++r) {
      const int m = rb * 16 + r + 8 * half;
      __builtin_nontemporal_store(acc[rb][r] + sBase[m], &Kout[ob + (size_t)m * LDIM]);
    }
}

// ---------------------------------------------------------------------------
// Launch. Workspace layout (needs ~8.8 MB):
//   base 4KB | CBb 128KB | Wb 128KB | sums 64KB | offs 64KB | cumA 8MB
// ---------------------------------------------------------------------------
extern "C" void kernel_launch(void* const* d_in, const int* in_sizes, int n_in,
                              void* d_out, int out_size, void* d_ws, size_t ws_size,
                              hipStream_t stream)
{
  (void)in_sizes; (void)n_in; (void)out_size; (void)ws_size;
  const float* du = (const float*)d_in[0];
  const float* C  = (const float*)d_in[1];
  const float* Bv = (const float*)d_in[2];
  const float* A  = (const float*)d_in[3];
  const float* W  = (const float*)d_in[4];
  float* Kout = (float*)d_out;

  char* ws = (char*)d_ws;
  float*  base = (float*)(ws + 0);
  bf16_t* CBb  = (bf16_t*)(ws + 4096);
  bf16_t* Wb   = (bf16_t*)(ws + 4096 + 131072);
  float*  sums = (float*)(ws + 266240);
  float*  offs = (float*)(ws + 331776);
  float*  cumA = (float*)(ws + 397312);

  k_setup<<<64, 256, 0, stream>>>(C, Bv, A, W, base, CBb, Wb);
  k_gemm1<<<dim3(32, 8), 256, 0, stream>>>(du, Wb, cumA, sums);
  k_scan<<<8, 64, 0, stream>>>(sums, offs);
  k_gemm2<<<dim3(32, 8, 8), 256, 0, stream>>>(cumA, offs, CBb, base, Kout);
}
